// ProposalGenerator_32006096289898
// MI455X (gfx1250) — matrix operand order repaired and checked by reading, not compiled, not run on hardware
//
#include <hip/hip_runtime.h>
#include <hip/hip_bf16.h>

// ---------------------------------------------------------------------------
// ProposalGenerator (BMN-style) for MI455X / gfx1250, wave32 + WMMA + TDM.
//
// Reordered contraction to avoid the 537MB "bm" intermediate:
//   x  = relu(conv1d(base, w1) + b1)              [8][256 t][128 c]  (bf16, ws)
//   z  = w3t[(o,k)][c] @ x  -> Zb[b][o][kt]       [8][512][8192]     (bf16, ws)
//   M' = bf16(mask) transposed -> Mb[hw][kt]      [4096][8192]       (bf16, ws, 64MB: L2-resident)
//   x3 = relu(Z_b @ M'^T + b3)  (275 GF bf16 WMMA, double-buffered TDM staging)
//   out= relu(w2 @ x3 + b2)     (fused second WMMA GEMM, epilogue)
//
// Workspace usage ~139 MB.
// ---------------------------------------------------------------------------

typedef __attribute__((ext_vector_type(16))) __bf16 bf16x16;
typedef __attribute__((ext_vector_type(8)))  __bf16 bf16x8;
typedef __attribute__((ext_vector_type(8)))  float  f32x8;
typedef __attribute__((ext_vector_type(4)))  unsigned int u32x4;
typedef __attribute__((ext_vector_type(8)))  int    i32x8;
typedef __attribute__((ext_vector_type(4)))  int    i32x4;

#define NB   8
#define C1   256
#define TT   256
#define C2   128
#define C3   512
#define NS   32
#define DD   64
#define HW   4096        // DD*DD
#define KT   8192        // NS*TT
#define KSTEPS (KT / 32) // 256 K-steps in the main GEMM

// LDS buffer byte offsets for k_bm_fused (computed arithmetically — no
// LDS-pointer tables, those trigger unsupported static initializers).
#define A_BYTES  (512 * 32 * 2)   // 32KB per A buffer
#define B_BYTES  (64 * 32 * 2)    // 4KB per B buffer
#define A_OFF    0u
#define B_OFF    (2u * A_BYTES)
#define XS_OFF   (2u * A_BYTES + 2u * B_BYTES)
#define SMEM_SZ  (XS_OFF + 64 * 512 * 2)          // + 64KB x3 tile = 139264 B

// ---- workspace layout (bytes) ----
#define MB_OFF   0ull                               // Mb  [HW][KT] bf16   64MB
#define ZB_OFF   (MB_OFF + (size_t)HW*KT*2)         // Zb  [B][512][KT]    64MB
#define X_OFF    (ZB_OFF + (size_t)NB*C3*KT*2)      // X   [B][T][C2] bf16 512KB
#define W3T_OFF  (X_OFF  + (size_t)NB*TT*C2*2)      // w3t [(o,k)][C2]     4MB

__device__ __forceinline__ bf16x16 frag_cat(bf16x8 lo, bf16x8 hi) {
  bf16x16 r;
#pragma unroll
  for (int i = 0; i < 8; ++i) { r[i] = lo[i]; r[i + 8] = hi[i]; }
  return r;
}

__device__ __forceinline__ f32x8 wmma_bf16(bf16x16 a, bf16x16 b, f32x8 c) {
  return __builtin_amdgcn_wmma_f32_16x16x32_bf16(false, a, false, b,
                                                 (short)0, c, false, false);
}

// Tensor Data Mover: 2-D tile of 2-byte elements, global -> LDS.
// D# per CDNA5 ISA ch.8: group0 = {flags, lds_addr, gaddr_lo, gaddr_hi|type},
// group1 packs data_size/tensor dims/tile dims/stride0.
__device__ __forceinline__ void tdm_load_2d_bf16(unsigned lds_off, const void* gptr,
                                                 unsigned tile_w, unsigned tile_h,
                                                 unsigned long long stride0_elems,
                                                 unsigned tensor_w, unsigned tensor_h) {
  unsigned long long ga = (unsigned long long)gptr;
  u32x4 g0;
  g0[0] = 1u;                                    // count=1, user descriptor
  g0[1] = lds_off;                               // LDS byte address
  g0[2] = (unsigned)(ga & 0xffffffffu);          // global_addr[31:0]
  g0[3] = (unsigned)((ga >> 32) & 0x01ffffffu) | (2u << 30);  // addr[56:32] | type=2
  i32x8 g1;
  g1[0] = (1 << 16);                                           // data_size=1 (2B)
  g1[1] = (int)((tensor_w & 0xffffu) << 16);                   // tensor_dim0 lo16
  g1[2] = (int)((tensor_w >> 16) | ((tensor_h & 0xffffu) << 16));
  g1[3] = (int)((tensor_h >> 16) | (tile_w << 16));            // tile_dim0
  g1[4] = (int)(tile_h & 0xffffu);                             // tile_dim1 (tile_dim2=0)
  g1[5] = (int)(stride0_elems & 0xffffffffu);                  // tensor_dim0_stride lo32
  g1[6] = (int)((stride0_elems >> 32) & 0xffffu);              // stride0 hi16 (stride1=0)
  g1[7] = 0;
  i32x4 z4 = {0, 0, 0, 0};
#if defined(__clang_major__) && (__clang_major__ >= 23)
  i32x8 z8 = {0, 0, 0, 0, 0, 0, 0, 0};
  __builtin_amdgcn_tensor_load_to_lds(g0, g1, z4, z4, z8, 0);
#else
  __builtin_amdgcn_tensor_load_to_lds(g0, g1, z4, z4, 0);
#endif
}

// ---------------------------------------------------------------------------
// K1: conv1d(k=3, pad=1) + bias + ReLU -> X[b][t][c2] bf16
// block = (b, c2), 256 threads over t. Weight loads are wave-uniform (SMEM).
// ---------------------------------------------------------------------------
__global__ __launch_bounds__(256) void k_conv1(const float* __restrict__ base,
                                               const float* __restrict__ w1,
                                               const float* __restrict__ b1,
                                               __bf16* __restrict__ X) {
  const int b  = blockIdx.x >> 7;
  const int c2 = blockIdx.x & 127;
  const int t  = threadIdx.x;
  const float* bf = base + (size_t)b * C1 * TT;
  const float* wp = w1 + (size_t)c2 * C1 * 3;
  float acc = b1[c2];
#pragma unroll 4
  for (int c1 = 0; c1 < C1; ++c1) {
    const float* r = bf + c1 * TT;
    float x0 = (t > 0)      ? r[t - 1] : 0.0f;
    float x1 = r[t];
    float x2 = (t < TT - 1) ? r[t + 1] : 0.0f;
    acc += x0 * wp[c1 * 3 + 0] + x1 * wp[c1 * 3 + 1] + x2 * wp[c1 * 3 + 2];
  }
  acc = fmaxf(acc, 0.0f);
  X[(size_t)b * TT * C2 + (size_t)t * C2 + c2] = (__bf16)acc;
}

// ---------------------------------------------------------------------------
// K2: w3[o][c][k] f32 -> w3t[(o*32+k)][c] bf16
// ---------------------------------------------------------------------------
__global__ __launch_bounds__(256) void k_w3t(const float* __restrict__ w3,
                                             __bf16* __restrict__ w3t) {
  const int dst = blockIdx.x * 256 + threadIdx.x;   // = o*4096 + k*128 + c
  const int o = dst >> 12, k = (dst >> 7) & 31, c = dst & 127;
  w3t[dst] = (__bf16)w3[(size_t)o * 4096 + (size_t)c * 32 + k];
}

// ---------------------------------------------------------------------------
// K3: mask[t][k*4096+hw] f32 -> Mb[hw][k*256+t] bf16 (LDS-tiled transpose)
// grid (hw/64, t/32, k), 256 threads.
// ---------------------------------------------------------------------------
__global__ __launch_bounds__(256) void k_mask(const float* __restrict__ mask,
                                              __bf16* __restrict__ Mb) {
  __shared__ float tile[32][65];
  const int h0 = blockIdx.x * 64, t0 = blockIdx.y * 32, k = blockIdx.z;
  const int tid = threadIdx.x;
#pragma unroll
  for (int e = tid; e < 2048; e += 256) {
    int r = e >> 6, c = e & 63;
    tile[r][c] = mask[(size_t)(t0 + r) * 131072 + (size_t)k * 4096 + h0 + c];
  }
  __syncthreads();
#pragma unroll
  for (int e = tid; e < 2048; e += 256) {
    int hh = e >> 5, tt = e & 31;
    Mb[(size_t)(h0 + hh) * KT + (size_t)k * 256 + t0 + tt] = (__bf16)tile[tt][hh];
  }
}

// ---------------------------------------------------------------------------
// K4: Zb[b] = w3t[16384 x 128] @ X_b[128 x 256], bf16 in / bf16 out.
// Block tile 128x64, 4 waves, each wave 2x4 WMMA tiles (64 accum VGPRs).
// ---------------------------------------------------------------------------
__global__ __launch_bounds__(128) void k_zgemm(const __bf16* __restrict__ w3t,
                                               const __bf16* __restrict__ X,
                                               __bf16* __restrict__ Zall) {
  const int tid = threadIdx.x, lane = tid & 31, w = tid >> 5;
  const int b = blockIdx.z, row0 = blockIdx.y * 128, col0 = blockIdx.x * 64;
  const __bf16* Xb = X + (size_t)b * TT * C2;          // [t][c]
  __bf16* Z = Zall + (size_t)b * C3 * KT;              // row (o*32+k), col t
  const int lh = lane >> 4, ll = lane & 15;

  f32x8 acc[2][4];
  f32x8 zf = {0.f, 0.f, 0.f, 0.f, 0.f, 0.f, 0.f, 0.f};
#pragma unroll
  for (int tr = 0; tr < 2; ++tr)
#pragma unroll
    for (int tc = 0; tc < 4; ++tc) acc[tr][tc] = zf;

#pragma unroll
  for (int ks = 0; ks < 4; ++ks) {                     // K = 128 = 4 x 32
    bf16x16 bfr[4];
#pragma unroll
    for (int tc = 0; tc < 4; ++tc) {
      int t = col0 + tc * 16 + ll;
      const __bf16* p = Xb + (size_t)t * C2 + ks * 32 + lh * 8;
      bfr[tc] = frag_cat(*(const bf16x8*)p, *(const bf16x8*)(p + 16));
    }
#pragma unroll
    for (int tr = 0; tr < 2; ++tr) {
      int row = row0 + w * 32 + tr * 16 + ll;
      const __bf16* p = w3t + (size_t)row * C2 + ks * 32;
      bf16x16 a = frag_cat(*(const bf16x8*)(p + lh * 8),
                           *(const bf16x8*)(p + 16 + lh * 8));
#pragma unroll
      for (int tc = 0; tc < 4; ++tc) acc[tr][tc] = wmma_bf16(a, bfr[tc], acc[tr][tc]);
    }
  }
#pragma unroll
  for (int tr = 0; tr < 2; ++tr)
#pragma unroll
    for (int tc = 0; tc < 4; ++tc)
#pragma unroll
      for (int r = 0; r < 8; ++r) {
        int m = row0 + w * 32 + tr * 16 + r + (lh << 3);
        int n = col0 + tc * 16 + ll;
        Z[(size_t)m * TT + n] = (__bf16)acc[tr][tc][r];
      }
}

// ---------------------------------------------------------------------------
// K5 (main): per block: b, 64-wide hw tile.
//   x3[512 x 64] = Z_b[512 x 8192] @ Mb[hw0..+64][8192]^T
//     256 K-steps, TDM-staged into DOUBLE-BUFFERED LDS tiles; the next step's
//     two tensor_load_to_lds ops are issued before waiting on the current
//     step's (TENSORcnt is per-wave and in-order, so s_wait_tensorcnt(2)
//     guarantees the older pair landed while the newer pair stays in flight).
//   LDS <- relu(x3 + b3) as bf16 [hw][o]
//   out[128 x 64] = relu(w2 @ x3 + b2)     (fused epilogue GEMM)
// 16 waves; wave grid 8x2 over [512 x 64]; 4x2 WMMA tiles per wave.
// LDS: 2*(32KB A + 4KB B) + 64KB x3 = 136KB -> 2 workgroups per WGP (320KB).
// ---------------------------------------------------------------------------
__global__ __launch_bounds__(512) void k_bm_fused(const __bf16* __restrict__ Zall,
                                                  const __bf16* __restrict__ Mb,
                                                  const float* __restrict__ b3,
                                                  const float* __restrict__ w2,
                                                  const float* __restrict__ b2,
                                                  float* __restrict__ out) {
  extern __shared__ char smem[];
  const unsigned lds_base = (unsigned)(size_t)smem;   // LDS byte addr (low 32b)
  __bf16* Xs = (__bf16*)(smem + XS_OFF);              // [64][512] x3 bf16

  const int tid = threadIdx.x, lane = tid & 31, w = tid >> 5;
  const int lh = lane >> 4, ll = lane & 15;
  const int b = blockIdx.y, hw0 = blockIdx.x * 64;
  const int wr = w & 7, wc = w >> 3;
  const __bf16* Zb = Zall + (size_t)b * C3 * KT;
  const __bf16* Mrow = Mb + (size_t)hw0 * KT;

  f32x8 acc[4][2];
  f32x8 zf = {0.f, 0.f, 0.f, 0.f, 0.f, 0.f, 0.f, 0.f};
#pragma unroll
  for (int tr = 0; tr < 4; ++tr)
#pragma unroll
    for (int tc = 0; tc < 2; ++tc) acc[tr][tc] = zf;

  // Prologue: stage K-step 0 into buffer 0 (one wave issues; TDM ignores EXEC).
  if (tid == 0) {
    tdm_load_2d_bf16(lds_base + A_OFF, Zb, 32, 512, KT, KT, 512);
    tdm_load_2d_bf16(lds_base + B_OFF, Mrow, 32, 64, KT, KT, HW);
  }

  for (int ks = 0; ks < KSTEPS; ++ks) {
    const unsigned cur = (unsigned)(ks & 1);
    const unsigned aoff = A_OFF + cur * A_BYTES;
    const unsigned boff = B_OFF + cur * B_BYTES;
    if (tid == 0) {
      if (ks + 1 < KSTEPS) {   // prefetch next K-step into the other buffer
        tdm_load_2d_bf16(lds_base + A_OFF + (cur ^ 1u) * A_BYTES,
                         Zb + (ks + 1) * 32, 32, 512, KT, KT, 512);
        tdm_load_2d_bf16(lds_base + B_OFF + (cur ^ 1u) * B_BYTES,
                         Mrow + (ks + 1) * 32, 32, 64, KT, KT, HW);
        __builtin_amdgcn_s_wait_tensorcnt(2);  // oldest pair (step ks) done
      } else {
        __builtin_amdgcn_s_wait_tensorcnt(0);  // drain for the final step
      }
    }
    __syncthreads();                           // publish buf[cur] to all waves

    const __bf16* As = (const __bf16*)(smem + aoff);
    const __bf16* Bs = (const __bf16*)(smem + boff);
    bf16x16 bfr[2];
#pragma unroll
    for (int tc = 0; tc < 2; ++tc) {
      int hl = wc * 32 + tc * 16 + ll;
      const __bf16* p = Bs + (size_t)hl * 32 + lh * 8;
      bfr[tc] = frag_cat(*(const bf16x8*)p, *(const bf16x8*)(p + 16));
    }
#pragma unroll
    for (int tr = 0; tr < 4; ++tr) {
      int row = wr * 64 + tr * 16 + ll;
      const __bf16* p = As + (size_t)row * 32;
      bf16x16 a = frag_cat(*(const bf16x8*)(p + lh * 8),
                           *(const bf16x8*)(p + 16 + lh * 8));
#pragma unroll
      for (int tc = 0; tc < 2; ++tc) acc[tr][tc] = wmma_bf16(a, bfr[tc], acc[tr][tc]);
    }
    __syncthreads();   // all waves done with buf[cur] before it is re-filled
  }

  // Epilogue 1: + b3, ReLU, park x3 in LDS as [hw][o] bf16 (B-operand layout).
#pragma unroll
  for (int tr = 0; tr < 4; ++tr)
#pragma unroll
    for (int tc = 0; tc < 2; ++tc)
#pragma unroll
      for (int r = 0; r < 8; ++r) {
        int o  = wr * 64 + tr * 16 + r + (lh << 3);
        int hl = wc * 32 + tc * 16 + ll;
        float v = acc[tr][tc][r] + b3[o];
        Xs[(size_t)hl * C3 + o] = (__bf16)fmaxf(v, 0.0f);
      }
  __syncthreads();

  // Epilogue 2: out[128 x 64] = relu(w2[128 x 512] @ x3 + b2).
  const int rt = w & 7, cg = w >> 3;
  f32x8 acc2[2] = {zf, zf};
#pragma unroll 4
  for (int ks2 = 0; ks2 < C3 / 32; ++ks2) {
    const float* wrow = w2 + (size_t)(rt * 16 + ll) * C3 + ks2 * 32;
    bf16x16 a2;
#pragma unroll
    for (int j = 0; j < 8; ++j) {
      a2[j]     = (__bf16)wrow[lh * 8 + j];
      a2[j + 8] = (__bf16)wrow[16 + lh * 8 + j];
    }
#pragma unroll
    for (int tc = 0; tc < 2; ++tc) {
      int hl = cg * 32 + tc * 16 + ll;
      const __bf16* p = Xs + (size_t)hl * C3 + ks2 * 32 + lh * 8;
      bf16x16 bv = frag_cat(*(const bf16x8*)p, *(const bf16x8*)(p + 16));
      acc2[tc] = wmma_bf16(a2, bv, acc2[tc]);
    }
  }
#pragma unroll
  for (int tc = 0; tc < 2; ++tc)
#pragma unroll
    for (int r = 0; r < 8; ++r) {
      int o2 = rt * 16 + r + (lh << 3);
      int hw = hw0 + cg * 32 + tc * 16 + ll;
      float v = acc2[tc][r] + b2[o2];
      out[((size_t)b * C2 + o2) * HW + hw] = fmaxf(v, 0.0f);
    }
}

// ---------------------------------------------------------------------------
extern "C" void kernel_launch(void* const* d_in, const int* in_sizes, int n_in,
                              void* d_out, int out_size, void* d_ws, size_t ws_size,
                              hipStream_t stream) {
  const float* base = (const float*)d_in[0];
  const float* w1   = (const float*)d_in[1];
  const float* b1   = (const float*)d_in[2];
  const float* w3   = (const float*)d_in[3];
  const float* b3   = (const float*)d_in[4];
  const float* w2   = (const float*)d_in[5];
  const float* b2   = (const float*)d_in[6];
  const float* mask = (const float*)d_in[7];
  float* out = (float*)d_out;

  char* ws = (char*)d_ws;
  __bf16* Mb  = (__bf16*)(ws + MB_OFF);
  __bf16* Zb  = (__bf16*)(ws + ZB_OFF);
  __bf16* X   = (__bf16*)(ws + X_OFF);
  __bf16* w3t = (__bf16*)(ws + W3T_OFF);

  k_conv1<<<NB * C2, 256, 0, stream>>>(base, w1, b1, X);
  k_w3t<<<(C3 * NS * C2) / 256, 256, 0, stream>>>(w3, w3t);
  k_mask<<<dim3(HW / 64, TT / 32, NS), 256, 0, stream>>>(mask, Mb);
  k_zgemm<<<dim3(TT / 64, (C3 * NS) / 128, NB), 128, 0, stream>>>(w3t, X, Zb);
  k_bm_fused<<<dim3(HW / 64, NB), 512, SMEM_SZ, stream>>>(Zb, Mb, b3, w2, b2, out);
}
